// SelfAttention_42855183679596
// MI455X (gfx1250) — matrix operand order repaired and checked by reading, not compiled, not run on hardware
//
#include <hip/hip_runtime.h>

#define BATCH 16
#define SEQ   2048
#define DIM   512

typedef _Float16 h16v __attribute__((ext_vector_type(16)));
typedef _Float16 h8v  __attribute__((ext_vector_type(8)));
typedef float    f8v  __attribute__((ext_vector_type(8)));

static __device__ __forceinline__ f8v wmma_f16(h16v a, h16v b, f8v c) {
    // D = A(16x32 f16) * B(32x16 f16) + C(16x16 f32)
    return __builtin_amdgcn_wmma_f32_16x16x32_f16(
        /*neg_a=*/false, a, /*neg_b=*/false, b,
        /*c_mod=*/(short)0, c, /*reuse_a=*/false, /*reuse_b=*/false);
}

static __device__ __forceinline__ f8v zero8() {
    f8v z = {0.f, 0.f, 0.f, 0.f, 0.f, 0.f, 0.f, 0.f};
    return z;
}

// -------------------------------------------------------------------------
// Kernel 1: fused QKV projection.  Q[b,s,e] = sum_d x[b,s,d] * W[e,d] + b[e]
// Each wave computes one 16(M) x 64(N) tile of one of {Q,K,V}.
// Q,K stored f16 row-major [B*S, D]; V stored f16 transposed [B, D, S].
// -------------------------------------------------------------------------
__global__ __launch_bounds__(256) void sa_qkv_kernel(
    const float* __restrict__ x,
    const float* __restrict__ Wq, const float* __restrict__ bq,
    const float* __restrict__ Wk, const float* __restrict__ bk,
    const float* __restrict__ Wv, const float* __restrict__ bv,
    _Float16* __restrict__ qh, _Float16* __restrict__ kh,
    _Float16* __restrict__ vt)
{
    const int lane = threadIdx.x & 31;
    const int nl   = lane & 15;
    const int half = lane >> 4;

    const int wg    = blockIdx.x * 8 + (threadIdx.x >> 5);
    const int n64   = wg & 7;          // which 64-wide column tile of DIM
    const int which = (wg >> 3) % 3;   // 0=Q 1=K 2=V
    const int mtile = wg / 24;         // 16-row tile over flattened B*S

    const float* W;
    const float* bias;
    if (which == 0)      { W = Wq; bias = bq; }
    else if (which == 1) { W = Wk; bias = bk; }
    else                 { W = Wv; bias = bv; }

    const long  row0 = (long)mtile * 16;            // flattened b*SEQ + s
    const float* xrow = x + (row0 + nl) * DIM;      // this lane's A row

    f8v acc[4];
    #pragma unroll
    for (int j = 0; j < 4; ++j) acc[j] = zero8();

    for (int k0 = 0; k0 < DIM; k0 += 32) {
        // ---- A fragment: 16x32 f16, row M = nl, K = k0 + half*16 + {0..15}
        const float4* pa0 = (const float4*)(xrow + k0 + half * 16);
        const float4* pa1 = (const float4*)(xrow + k0 + half * 16 + 8);
        float4 a0 = pa0[0], a1 = pa0[1];
        float4 a2 = pa1[0], a3 = pa1[1];
        h16v af;
        af[0]  = (_Float16)a0.x; af[1]  = (_Float16)a0.y;
        af[2]  = (_Float16)a0.z; af[3]  = (_Float16)a0.w;
        af[4]  = (_Float16)a1.x; af[5]  = (_Float16)a1.y;
        af[6]  = (_Float16)a1.z; af[7]  = (_Float16)a1.w;
        af[8]  = (_Float16)a2.x; af[9]  = (_Float16)a2.y;
        af[10] = (_Float16)a2.z; af[11] = (_Float16)a2.w;
        af[12] = (_Float16)a3.x; af[13] = (_Float16)a3.y;
        af[14] = (_Float16)a3.z; af[15] = (_Float16)a3.w;

        #pragma unroll
        for (int j = 0; j < 4; ++j) {
            // ---- B fragment: B[k][n] = W[n][k] -> lane's column n reads W row
            const int nrow = n64 * 64 + j * 16 + nl;
            const float4* pb = (const float4*)(W + (long)nrow * DIM + k0 + 16 * half);
            float4 b0 = pb[0], b1 = pb[1], b2 = pb[2], b3 = pb[3];
            h16v bf;
            bf[0]  = (_Float16)b0.x; bf[1]  = (_Float16)b0.y;
            bf[2]  = (_Float16)b0.z; bf[3]  = (_Float16)b0.w;
            bf[4]  = (_Float16)b1.x; bf[5]  = (_Float16)b1.y;
            bf[6]  = (_Float16)b1.z; bf[7]  = (_Float16)b1.w;
            bf[8]  = (_Float16)b2.x; bf[9]  = (_Float16)b2.y;
            bf[10] = (_Float16)b2.z; bf[11] = (_Float16)b2.w;
            bf[12] = (_Float16)b3.x; bf[13] = (_Float16)b3.y;
            bf[14] = (_Float16)b3.z; bf[15] = (_Float16)b3.w;
            acc[j] = wmma_f16(af, bf, acc[j]);
        }
    }

    // ---- epilogue: add bias, store f16.  C/D layout: lane holds (M=8*half+v, N=nl)
    const long bnum = row0 / SEQ;
    const int  sloc = (int)(row0 % SEQ);
    #pragma unroll
    for (int j = 0; j < 4; ++j) {
        const int  nglob = n64 * 64 + j * 16 + nl;
        const float bb   = bias[nglob];
        if (which < 2) {
            _Float16* dst = (which == 0) ? qh : kh;
            #pragma unroll
            for (int v = 0; v < 8; ++v) {
                const int m = half * 8 + v;
                dst[(row0 + m) * DIM + nglob] = (_Float16)(acc[j][v] + bb);
            }
        } else {
            h8v pk;
            #pragma unroll
            for (int v = 0; v < 8; ++v) pk[v] = (_Float16)(acc[j][v] + bb);
            // Vt[b][e][s]: consecutive v are consecutive s -> one 16B store
            *(h8v*)(vt + ((long)bnum * DIM + nglob) * SEQ + sloc + half * 8) = pk;
        }
    }
}

// -------------------------------------------------------------------------
// Kernel 2: scores + softmax + weights output + out = P*V.
// One block (16 waves) per 16-query strip; wave w owns key cols [w*128,w*128+128).
// -------------------------------------------------------------------------
__global__ __launch_bounds__(512) void sa_attn_kernel(
    const _Float16* __restrict__ qh, const _Float16* __restrict__ kh,
    const _Float16* __restrict__ vt,
    float* __restrict__ outp, float* __restrict__ wout)
{
    __shared__ alignas(32) _Float16 p_lds[16][SEQ];   // 64 KB probability strip
    __shared__ float red[16][16];                     // per-wave row partials
    __shared__ float rowstat[16];                     // reduced per-row stat

    const int lane = threadIdx.x & 31;
    const int nl   = lane & 15;
    const int half = lane >> 4;
    const int wv   = threadIdx.x >> 5;

    const int b     = blockIdx.x / (SEQ / 16);
    const int qbase = (blockIdx.x % (SEQ / 16)) * 16;

    const _Float16* qrow = qh + ((long)b * SEQ + qbase + nl) * DIM;

    // ---------------- Phase A: scores = Q * K^T (raw, unscaled) ------------
    f8v acc[8];
    #pragma unroll
    for (int t = 0; t < 8; ++t) acc[t] = zero8();

    for (int d0 = 0; d0 < DIM; d0 += 32) {
        h8v c0 = *(const h8v*)(qrow + d0 + half * 16);
        h8v c1 = *(const h8v*)(qrow + d0 + half * 16 + 8);
        h16v af;
        #pragma unroll
        for (int i = 0; i < 8; ++i) { af[i] = c0[i]; af[8 + i] = c1[i]; }
        #pragma unroll
        for (int t = 0; t < 8; ++t) {
            const int key = wv * 128 + t * 16 + nl;      // B col n = key row of K
            h16v bf = *(const h16v*)(kh + ((long)b * SEQ + key) * DIM + d0 + 16 * half);
            acc[t] = wmma_f16(af, bf, acc[t]);
        }
    }

    const float scale = 0.044194173824159216f;  // 1/sqrt(512)

    // ---------------- row max (over 2048 cols) -----------------------------
    float r8[8];
    #pragma unroll
    for (int v = 0; v < 8; ++v) {
        float m = acc[0][v];
        #pragma unroll
        for (int t = 1; t < 8; ++t) m = fmaxf(m, acc[t][v]);
        r8[v] = m;
    }
    for (int s = 1; s < 16; s <<= 1) {
        #pragma unroll
        for (int v = 0; v < 8; ++v)
            r8[v] = fmaxf(r8[v], __shfl_xor(r8[v], s, 32));
    }
    if (nl == 0) {
        #pragma unroll
        for (int v = 0; v < 8; ++v) red[wv][half * 8 + v] = r8[v];
    }
    __syncthreads();
    if (wv == 0 && lane < 16) {
        float g = red[0][lane];
        for (int w2 = 1; w2 < 16; ++w2) g = fmaxf(g, red[w2][lane]);
        rowstat[lane] = g;
    }
    __syncthreads();

    float gmax[8], rsum[8];
    #pragma unroll
    for (int v = 0; v < 8; ++v) { gmax[v] = rowstat[half * 8 + v]; rsum[v] = 0.f; }

    // ---------------- exp + row sum ----------------------------------------
    #pragma unroll
    for (int t = 0; t < 8; ++t) {
        #pragma unroll
        for (int v = 0; v < 8; ++v) {
            float p = __expf((acc[t][v] - gmax[v]) * scale);
            acc[t][v] = p;
            rsum[v] += p;
        }
    }
    for (int s = 1; s < 16; s <<= 1) {
        #pragma unroll
        for (int v = 0; v < 8; ++v)
            rsum[v] += __shfl_xor(rsum[v], s, 32);
    }
    if (nl == 0) {
        #pragma unroll
        for (int v = 0; v < 8; ++v) red[wv][half * 8 + v] = rsum[v];
    }
    __syncthreads();
    if (wv == 0 && lane < 16) {
        float g = red[0][lane];
        for (int w2 = 1; w2 < 16; ++w2) g += red[w2][lane];
        rowstat[lane] = g;
    }
    __syncthreads();

    float inv[8];
    #pragma unroll
    for (int v = 0; v < 8; ++v) inv[v] = 1.0f / rowstat[half * 8 + v];

    // ---------------- write weights (f32 out) + f16 probs to LDS -----------
    #pragma unroll
    for (int t = 0; t < 8; ++t) {
        const int col = wv * 128 + t * 16 + nl;
        #pragma unroll
        for (int v = 0; v < 8; ++v) {
            const int m = half * 8 + v;
            const float pn = acc[t][v] * inv[v];
            wout[((long)b * SEQ + qbase + m) * SEQ + col] = pn;
            p_lds[m][col] = (_Float16)pn;
        }
    }
    __syncthreads();

    // ---------------- Phase C: out = P * V  (wave owns e in [wv*32,wv*32+32))
    f8v o0 = zero8(), o1 = zero8();
    const int e0 = wv * 32;
    const _Float16* vrow0 = vt + ((long)b * DIM + e0 + nl) * SEQ;
    const _Float16* vrow1 = vt + ((long)b * DIM + e0 + 16 + nl) * SEQ;

    for (int k0 = 0; k0 < SEQ; k0 += 32) {
        h8v c0 = *(const h8v*)(&p_lds[nl][k0 + half * 16]);
        h8v c1 = *(const h8v*)(&p_lds[nl][k0 + half * 16 + 8]);
        h16v af;
        #pragma unroll
        for (int i = 0; i < 8; ++i) { af[i] = c0[i]; af[8 + i] = c1[i]; }
        h16v b0 = *(const h16v*)(vrow0 + k0 + 16 * half);
        h16v b1 = *(const h16v*)(vrow1 + k0 + 16 * half);
        o0 = wmma_f16(af, b0, o0);
        o1 = wmma_f16(af, b1, o1);
    }
    #pragma unroll
    for (int v = 0; v < 8; ++v) {
        const int m = half * 8 + v;
        const long obase = ((long)b * SEQ + qbase + m) * DIM;
        outp[obase + e0 + nl]      = o0[v];
        outp[obase + e0 + 16 + nl] = o1[v];
    }
}

// -------------------------------------------------------------------------
extern "C" void kernel_launch(void* const* d_in, const int* in_sizes, int n_in,
                              void* d_out, int out_size, void* d_ws, size_t ws_size,
                              hipStream_t stream) {
    const float* x  = (const float*)d_in[0];
    const float* Wq = (const float*)d_in[1];
    const float* bq = (const float*)d_in[2];
    const float* Wk = (const float*)d_in[3];
    const float* bk = (const float*)d_in[4];
    const float* Wv = (const float*)d_in[5];
    const float* bv = (const float*)d_in[6];

    float* outp = (float*)d_out;                                   // [B,S,D]
    float* wout = outp + (size_t)BATCH * SEQ * DIM;                // [B,S,S]

    const size_t elems = (size_t)BATCH * SEQ * DIM;                // 16.8M
    _Float16* qh = (_Float16*)d_ws;                                // 32 MB
    _Float16* kh = qh + elems;                                     // 32 MB
    _Float16* vt = kh + elems;                                     // 32 MB (transposed V)

    // 3 weights * 2048 M-tiles * 8 N64-tiles = 49152 wave-tasks / 8 waves per block
    sa_qkv_kernel<<<6144, 256, 0, stream>>>(x, Wq, bq, Wk, bk, Wv, bv, qh, kh, vt);
    // one block per 16-query strip: 16 * (2048/16) = 2048 blocks, 16 waves each
    sa_attn_kernel<<<2048, 512, 0, stream>>>(qh, kh, vt, outp, wout);
}
